// DAWN_26259430048186
// MI455X (gfx1250) — matrix operand order repaired and checked by reading, not compiled, not run on hardware
//
#include <hip/hip_runtime.h>
#include <math.h>

// ---------------- problem constants ----------------
#define Bsz   4
#define Sq    1024
#define Dm    512
#define Hh    8
#define Dh    64
#define Ll    4
#define Mtok  (Bsz*Sq)      // 4096 tokens
#define NB_   32
#define RANK_ 64
#define NNEU_ 64
#define DFF_  2048
#define VOCAB_ 32000
#define KSEL_ 8

typedef __attribute__((ext_vector_type(16))) __bf16 v16bf;
typedef __attribute__((ext_vector_type(8)))  float  v8f;
typedef unsigned int u32x4 __attribute__((ext_vector_type(4)));
typedef int          i32x4 __attribute__((ext_vector_type(4)));
typedef int          i32x8 __attribute__((ext_vector_type(8)));

__device__ __forceinline__ unsigned short f2bf(float f) {
  union { float f; unsigned int u; } c; c.f = f;
  unsigned int u = c.u;
  return (unsigned short)((u + 0x7FFFu + ((u >> 16) & 1u)) >> 16); // RNE
}

__device__ __forceinline__ v8f v8f_zero() {
  v8f z;
#pragma unroll
  for (int i = 0; i < 8; ++i) z[i] = 0.f;
  return z;
}

// ---- Tensor Data Mover: async 2D bf16 tile (tileX x tileY elems) -> LDS ----
// D# per CDNA5 ISA ch.8: group0 = {count|flags, lds_addr, global_addr, type},
// group1 = {mask/data_size, tensor dims, tile dims, strides}. 2D: groups 2/3 zero.
__device__ __forceinline__ void tdm_load_2d(unsigned lds_off,
                                            const unsigned short* gptr,
                                            unsigned tileX, unsigned tileY,
                                            unsigned dimX, unsigned dimY,
                                            unsigned long long strideX /*elems*/) {
  unsigned long long ga = (unsigned long long)(size_t)gptr;
  u32x4 g0;
  g0[0] = 1u;                                         // count=1, user descriptor
  g0[1] = lds_off;                                    // lds_addr (bytes)
  g0[2] = (unsigned)(ga & 0xFFFFFFFFull);             // global_addr[31:0]
  g0[3] = (unsigned)((ga >> 32) & 0x1FFFFFFull) | (2u << 30); // addr[56:32]|type=2
  i32x8 g1;
  g1[0] = (int)(1u << 16);                            // data_size=1 (2 bytes)
  g1[1] = (int)((dimX & 0xFFFFu) << 16);              // tensor_dim0[15:0]
  g1[2] = (int)((dimX >> 16) | ((dimY & 0xFFFFu) << 16));   // td0[31:16]|td1[15:0]
  g1[3] = (int)((dimY >> 16) | (tileX << 16));        // td1[31:16]|tile_dim0
  g1[4] = (int)(tileY & 0xFFFFu);                     // tile_dim1 (tile_dim2=0)
  g1[5] = (int)(unsigned)(strideX & 0xFFFFFFFFull);   // tensor_dim0_stride[31:0]
  g1[6] = (int)(unsigned)((strideX >> 32) & 0xFFFFull); // stride[47:32]
  g1[7] = 0;
  i32x4 z4 = {0, 0, 0, 0};
#if defined(__clang_major__) && (__clang_major__ >= 23)
  i32x8 z8 = {0, 0, 0, 0, 0, 0, 0, 0};
  __builtin_amdgcn_tensor_load_to_lds(g0, g1, z4, z4, z8, 0);
#else
  __builtin_amdgcn_tensor_load_to_lds(g0, g1, z4, z4, 0);
#endif
}

// ---------------- embedding ----------------
__global__ __launch_bounds__(256)
void embed_kernel(const int* __restrict__ ids, const float* __restrict__ te,
                  const float* __restrict__ pe, float* __restrict__ x) {
  int i = blockIdx.x * 256 + threadIdx.x;
  if (i >= Mtok * Dm) return;
  int tok = i >> 9, d = i & (Dm - 1);
  int s = tok & (Sq - 1);
  int id = ids[tok];
  x[i] = te[(size_t)id * Dm + d] + pe[(size_t)s * Dm + d];
}

// ---------------- weight prep: f32 [K,N] -> bf16 [N,K] (transpose) ----------
__global__ __launch_bounds__(256)
void convT_kernel(const float* __restrict__ src, unsigned short* __restrict__ dst,
                  int K, int N) {
  int i = blockIdx.x * 256 + threadIdx.x;
  if (i >= K * N) return;
  int n = i / K, k = i - n * K;
  dst[i] = f2bf(src[(size_t)k * N + n]);
}

// f32 -> bf16 straight copy (token_emb is already [N,K])
__global__ __launch_bounds__(256)
void conv_kernel(const float* __restrict__ src, unsigned short* __restrict__ dst,
                 int cnt) {
  int i = blockIdx.x * 256 + threadIdx.x;
  if (i < cnt) dst[i] = f2bf(src[i]);
}

// basis_A [NB,D,R] -> bf16 [NB*R, D]  (B^T layout for the t-GEMM)
__global__ __launch_bounds__(256)
void packAT_kernel(const float* __restrict__ bA, unsigned short* __restrict__ pAT) {
  int i = blockIdx.x * 256 + threadIdx.x;           // dst linear over [NB*R, D]
  if (i >= NB_ * RANK_ * Dm) return;
  int d = i & (Dm - 1);
  int nr = i >> 9;
  int n = nr >> 6, r = nr & (RANK_ - 1);
  pAT[i] = f2bf(bA[((size_t)n * Dm + d) * RANK_ + r]);
}

// ---------------- layernorm (row of 512) ----------------
__global__ __launch_bounds__(128)
void ln_kernel(const float* __restrict__ x, const float* __restrict__ w,
               const float* __restrict__ b, float* __restrict__ out, int ldo) {
  int row = blockIdx.x;
  const float* xr = x + (size_t)row * Dm;
  __shared__ float red[128];
  int tid = threadIdx.x;
  float s = 0.f;
  for (int i = tid; i < Dm; i += 128) s += xr[i];
  red[tid] = s; __syncthreads();
  for (int o = 64; o > 0; o >>= 1) { if (tid < o) red[tid] += red[tid + o]; __syncthreads(); }
  float mu = red[0] * (1.f / Dm);
  __syncthreads();
  float vs = 0.f;
  for (int i = tid; i < Dm; i += 128) { float d = xr[i] - mu; vs += d * d; }
  red[tid] = vs; __syncthreads();
  for (int o = 64; o > 0; o >>= 1) { if (tid < o) red[tid] += red[tid + o]; __syncthreads(); }
  float inv = rsqrtf(red[0] * (1.f / Dm) + 1e-5f);
  for (int i = tid; i < Dm; i += 128)
    out[(size_t)row * ldo + i] = (xr[i] - mu) * inv * w[i] + b[i];
}

// -------- neuron_emb = softmax(recipe_l) @ basis_emb  (bf16 [N=64,K=512]) ----
__global__ __launch_bounds__(128)
void nemb_kernel(const float* __restrict__ recipe_l, const float* __restrict__ be,
                 unsigned short* __restrict__ nemb) {
  int n = blockIdx.x;                       // 0..63
  const float* rr = recipe_l + n * NB_;
  float mx = -3e38f;
#pragma unroll
  for (int j = 0; j < NB_; ++j) mx = fmaxf(mx, rr[j]);
  float e[NB_]; float se = 0.f;
#pragma unroll
  for (int j = 0; j < NB_; ++j) { e[j] = __expf(rr[j] - mx); se += e[j]; }
  float inv = 1.f / se;
  for (int d = threadIdx.x; d < Dm; d += 128) {
    float a = 0.f;
#pragma unroll
    for (int j = 0; j < NB_; ++j) a += e[j] * be[(size_t)j * Dm + d];
    nemb[(size_t)n * Dm + d] = f2bf(a * inv);
  }
}

// ---------------- router: top-8 of 64 + softmax, mix recipe rows -------------
__global__ __launch_bounds__(256)
void router_kernel(const float* __restrict__ scores, const float* __restrict__ recipe_l,
                   float* __restrict__ tr, int M) {
  int t = blockIdx.x * 256 + threadIdx.x;
  if (t >= M) return;
  const float* sc = scores + (size_t)t * NNEU_;
  unsigned long long selmask = 0ull;
  int idx[KSEL_]; float val[KSEL_];
#pragma unroll
  for (int kk = 0; kk < KSEL_; ++kk) {
    int bi = 0; float bv = -3e38f;
    for (int i = 0; i < NNEU_; ++i) {
      float xv = sc[i];
      if (!((selmask >> i) & 1ull) && xv > bv) { bv = xv; bi = i; }
    }
    selmask |= (1ull << bi);
    idx[kk] = bi; val[kk] = bv;
  }
  float m0 = val[0], se = 0.f, w[KSEL_];
#pragma unroll
  for (int kk = 0; kk < KSEL_; ++kk) { w[kk] = __expf(val[kk] - m0); se += w[kk]; }
  float acc[NB_];
#pragma unroll
  for (int n = 0; n < NB_; ++n) acc[n] = 0.f;
#pragma unroll
  for (int kk = 0; kk < KSEL_; ++kk) {
    const float* rr = recipe_l + (size_t)idx[kk] * NB_;
    float rm = -3e38f;
#pragma unroll
    for (int n = 0; n < NB_; ++n) rm = fmaxf(rm, rr[n]);
    float e[NB_]; float rs = 0.f;
#pragma unroll
    for (int n = 0; n < NB_; ++n) { e[n] = __expf(rr[n] - rm); rs += e[n]; }
    float wk = (w[kk] / se) / rs;
#pragma unroll
    for (int n = 0; n < NB_; ++n) acc[n] += wk * e[n];
  }
#pragma unroll
  for (int n = 0; n < NB_; ++n) tr[(size_t)t * NB_ + n] = acc[n];
}

// ---------------- h = tr . t ; u = tr (x) h  (in place on t) ----------------
__global__ __launch_bounds__(64)
void mix_kernel(float* __restrict__ t, const float* __restrict__ tr) {
  int tok = blockIdx.x;
  int r = threadIdx.x;                      // 0..63
  __shared__ float trs[NB_];
  if (r < NB_) trs[r] = tr[(size_t)tok * NB_ + r];
  __syncthreads();
  float* tp = t + (size_t)tok * DFF_;
  float h = 0.f;
#pragma unroll
  for (int n = 0; n < NB_; ++n) h += trs[n] * tp[n * RANK_ + r];
#pragma unroll
  for (int n = 0; n < NB_; ++n) tp[n * RANK_ + r] = trs[n] * h;
}

// ------- bf16 WMMA GEMM with TDM-fed, double-buffered B tiles ---------------
// C[M,N] = act(A[M,K](f32) @ Bw^T + bias) (+resid).  Bw: bf16 [N,K] row-major.
// ACT: 0 none, 1 exact GELU.  RES: C += resid (row stride N).
template <int ACT, bool RES>
__global__ __launch_bounds__(256)
void gemm_tdm(const float* __restrict__ A, int lda,
              const unsigned short* __restrict__ Bw,
              const float* __restrict__ bias,
              const float* resid, float* C,
              int M, int N, int K) {
  const int BM = 128, BN = 128, BK = 32;
  __shared__ unsigned short As[BM][BK];        // [row][k]
  __shared__ unsigned short Bs[2][BN][BK];     // [n][k], TDM double buffer
  int bm = blockIdx.y * BM;
  int bn = blockIdx.x * BN;
  int tid = threadIdx.x;
  int lane = tid & 31, wave = tid >> 5;
  int wm = wave & 3;          // 4 waves along M (32 rows each)
  int wn = wave >> 2;         // 2 waves along N (64 cols each)
  int half = lane >> 4, l16 = lane & 15;

  const unsigned short* gB = Bw + (size_t)bn * K;   // tile row base [N,K]
  unsigned dimY = (unsigned)(N - bn);               // TDM zero-fills rows >= dimY
  int ksteps = K / BK;

  // prologue: async-fetch B tile 0
  if (wave == 0)
    tdm_load_2d((unsigned)(size_t)&Bs[0][0][0], gB, BK, BN, (unsigned)K, dimY,
                (unsigned long long)K);

  v8f acc[2][4];
#pragma unroll
  for (int a = 0; a < 2; ++a)
#pragma unroll
    for (int c = 0; c < 4; ++c) acc[a][c] = v8f_zero();

  for (int it = 0; it < ksteps; ++it) {
    int k0 = it * BK;
    // stage A tile (f32 -> bf16), prefetch next K-step
    for (int i = tid; i < BM * BK; i += 256) {
      int r = i >> 5, c = i & 31;
      int gr = bm + r, gc = k0 + c;
      const float* ap = &A[(size_t)gr * lda + gc];
      float v = (gr < M) ? *ap : 0.f;
      As[r][c] = f2bf(v);
      if (it + 1 < ksteps) __builtin_prefetch(ap + BK, 0, 3);
    }
    // async-fetch next B tile, then wait for current one
    if (wave == 0) {
      if (it + 1 < ksteps) {
        tdm_load_2d((unsigned)(size_t)&Bs[(it + 1) & 1][0][0], gB + (k0 + BK),
                    BK, BN, (unsigned)(K - (k0 + BK)), dimY,
                    (unsigned long long)K);
        __builtin_amdgcn_s_wait_tensorcnt(1);   // current tile complete (in-order)
      } else {
        __builtin_amdgcn_s_wait_tensorcnt(0);
      }
    }
    __syncthreads();

    const unsigned short(*Bcur)[BK] = Bs[it & 1];
    // B fragments (shared across the two M sub-tiles). lane: n=l16, k=half*16+e
    v16bf bf[4];
#pragma unroll
    for (int nf = 0; nf < 4; ++nf) {
      union { v16bf v; unsigned int u[8]; } t;
      const unsigned int* p32 = (const unsigned int*)&Bcur[wn * 64 + nf * 16 + l16][half * 16];
#pragma unroll
      for (int j = 0; j < 8; ++j) t.u[j] = p32[j];
      bf[nf] = t.v;
    }
#pragma unroll
    for (int mf = 0; mf < 2; ++mf) {
      // A fragment. lane: row=l16, k = half*16 + vv*2 + {0,1}
      union { v16bf v; unsigned int u[8]; } t;
      const unsigned int* p32 = (const unsigned int*)&As[wm * 32 + mf * 16 + l16][0];
#pragma unroll
      for (int j = 0; j < 8; ++j) {
        int kk = half * 16 + j * 2;
        t.u[j] = p32[kk >> 1];
      }
      v16bf af = t.v;
#pragma unroll
      for (int nf = 0; nf < 4; ++nf)
        acc[mf][nf] = __builtin_amdgcn_wmma_f32_16x16x32_bf16(
            false, af, false, bf[nf], (short)0, acc[mf][nf], false, false);
    }
    __syncthreads();
  }

  // epilogue: lane holds (m = half*8+i, n = l16) of each 16x16 frag
#pragma unroll
  for (int mf = 0; mf < 2; ++mf)
#pragma unroll
    for (int nf = 0; nf < 4; ++nf)
#pragma unroll
      for (int i = 0; i < 8; ++i) {
        int m = bm + wm * 32 + mf * 16 + half * 8 + i;
        int n = bn + wn * 64 + nf * 16 + l16;
        if (m < M && n < N) {
          float v = acc[mf][nf][i];
          if (bias) v += bias[n];
          if (ACT == 1) v = 0.5f * v * (1.f + erff(v * 0.70710678118654752f));
          if (RES) v += resid[(size_t)m * N + n];
          C[(size_t)m * N + n] = v;
        }
      }
}

// ---------------- fused causal flash attention (WMMA bf16) ----------------
// grid = (S/64, B*H), block = 128 (4 waves x 16 queries). ctx row stride ldc.
__global__ __launch_bounds__(128)
void attn_kernel(const float* __restrict__ q, const float* __restrict__ k,
                 const float* __restrict__ v, float* ctx, int ldc) {
  int bh = blockIdx.y;
  int b = bh >> 3, h = bh & 7;
  int q0blk = blockIdx.x * 64;
  int tid = threadIdx.x, lane = tid & 31, wave = tid >> 5;
  int half = lane >> 4, l16 = lane & 15;
  int qrow0 = q0blk + wave * 16;

  const float* qbase = q + (size_t)b * Sq * Dm + h * Dh;
  const float* kbase = k + (size_t)b * Sq * Dm + h * Dh;
  const float* vbase = v + (size_t)b * Sq * Dm + h * Dh;

  __shared__ unsigned short Ks[32][64];       // [key][dh]
  __shared__ unsigned short Vs[64][32];       // [dh][key] (transposed)
  __shared__ unsigned short Ps[4][16][32];    // per-wave P tile

  // Q A-fragments (dh=64 -> two K=32 slices); 1/sqrt(dh) folded into Q.
  v16bf qa[2];
  {
    const float* qr = qbase + (size_t)(qrow0 + l16) * Dm;
#pragma unroll
    for (int ks2 = 0; ks2 < 2; ++ks2) {
      union { v16bf v; unsigned short s[16]; } t;
#pragma unroll
      for (int e = 0; e < 16; ++e) {
        int j = e >> 1, h2 = e & 1;
        int kk = ks2 * 32 + half * 16 + e;
        t.s[e] = f2bf(qr[kk] * 0.125f);
      }
      qa[ks2] = t.v;
    }
  }

  v8f acc[4];
#pragma unroll
  for (int nf = 0; nf < 4; ++nf) acc[nf] = v8f_zero();
  float m_arr[8], l_arr[8];
#pragma unroll
  for (int i = 0; i < 8; ++i) { m_arr[i] = -1e30f; l_arr[i] = 0.f; }

  int kbmax = (q0blk + 63) >> 5;        // block-uniform loop bound
  int kbact = (qrow0 + 15) >> 5;        // last block with any live key (per wave)
  for (int kblk = 0; kblk <= kbmax; ++kblk) {
    for (int i = tid; i < 32 * 64; i += 128) {
      int key = i >> 6, d = i & 63;
      size_t g = (size_t)(kblk * 32 + key) * Dm + d;
      Ks[key][d] = f2bf(kbase[g]);
      Vs[d][key] = f2bf(vbase[g]);
    }
    __syncthreads();

    if (kblk <= kbact) {   // wave-uniform causal skip; EXEC stays all-ones
      // S = Q K^T : two 16-key column tiles, K-dim 64 = 2 wmmas each
      v8f sc[2];
#pragma unroll
      for (int ct = 0; ct < 2; ++ct) {
        v8f s = v8f_zero();
#pragma unroll
        for (int ks2 = 0; ks2 < 2; ++ks2) {
          union { v16bf v; unsigned int u[8]; } t;
          const unsigned int* p32 =
              (const unsigned int*)&Ks[ct * 16 + l16][ks2 * 32 + half * 16];
#pragma unroll
          for (int j = 0; j < 8; ++j) t.u[j] = p32[j];
          s = __builtin_amdgcn_wmma_f32_16x16x32_bf16(
              false, qa[ks2], false, t.v, (short)0, s, false, false);
        }
        sc[ct] = s;
      }

      // online softmax update (row stats replicated across each 16-lane half)
      float p0[8], p1[8];
#pragma unroll
      for (int i = 0; i < 8; ++i) {
        int qrow = qrow0 + half * 8 + i;
        int key0 = kblk * 32 + l16;
        int key1 = key0 + 16;
        float a0 = (key0 <= qrow) ? sc[0][i] : -1e30f;
        float a1 = (key1 <= qrow) ? sc[1][i] : -1e30f;
        float rm = fmaxf(a0, a1);
        for (int o = 1; o < 16; o <<= 1) rm = fmaxf(rm, __shfl_xor(rm, o, 32));
        float nm = fmaxf(m_arr[i], rm);
        float scl = __expf(m_arr[i] - nm);
        float e0 = __expf(a0 - nm);
        float e1 = __expf(a1 - nm);
        float rs = e0 + e1;
        for (int o = 1; o < 16; o <<= 1) rs += __shfl_xor(rs, o, 32);
        l_arr[i] = l_arr[i] * scl + rs;
        m_arr[i] = nm;
        p0[i] = e0; p1[i] = e1;
#pragma unroll
        for (int nf = 0; nf < 4; ++nf) acc[nf][i] *= scl;
      }

      // stage P (C-layout) into LDS, regather as A fragment [16 q x 32 keys]
#pragma unroll
      for (int i = 0; i < 8; ++i) {
        Ps[wave][half * 8 + i][l16]      = f2bf(p0[i]);
        Ps[wave][half * 8 + i][16 + l16] = f2bf(p1[i]);
      }
      v16bf pf;
      {
        union { v16bf v; unsigned int u[8]; } t;
        const unsigned int* p32 = (const unsigned int*)&Ps[wave][l16][0];
#pragma unroll
        for (int j = 0; j < 8; ++j) {
          int kk = half * 16 + j * 2;
          t.u[j] = p32[kk >> 1];
        }
        pf = t.v;
      }
      // O += P @ V  (4 dh-column tiles)
#pragma unroll
      for (int nf = 0; nf < 4; ++nf) {
        union { v16bf v; unsigned int u[8]; } t;
        const unsigned int* p32 = (const unsigned int*)&Vs[nf * 16 + l16][half * 16];
#pragma unroll
        for (int j = 0; j < 8; ++j) t.u[j] = p32[j];
        acc[nf] = __builtin_amdgcn_wmma_f32_16x16x32_bf16(
            false, pf, false, t.v, (short)0, acc[nf], false, false);
      }
    }
    __syncthreads();
  }

#pragma unroll
  for (int nf = 0; nf < 4; ++nf)
#pragma unroll
    for (int i = 0; i < 8; ++i) {
      int qrow = qrow0 + half * 8 + i;
      ctx[(size_t)(b * Sq + qrow) * ldc + h * Dh + nf * 16 + l16] =
          acc[nf][i] / l_arr[i];
    }
}

// ---------------- host orchestration ----------------
extern "C" void kernel_launch(void* const* d_in, const int* in_sizes, int n_in,
                              void* d_out, int out_size, void* d_ws, size_t ws_size,
                              hipStream_t stream) {
  const int*   input_ids = (const int*)d_in[0];
  const float* token_emb = (const float*)d_in[1];
  const float* pos_emb   = (const float*)d_in[2];
  const float* basis_A   = (const float*)d_in[3];
  const float* basis_B   = (const float*)d_in[4];
  const float* basis_emb = (const float*)d_in[5];
  const float* recipe    = (const float*)d_in[6];
  const float* Wq = (const float*)d_in[7];
  const float* bq = (const float*)d_in[8];
  const float* Wk = (const float*)d_in[9];
  const float* bk = (const float*)d_in[10];
  const float* Wv = (const float*)d_in[11];
  const float* bv = (const float*)d_in[12];
  const float* Ws = (const float*)d_in[13];
  const float* bscore = (const float*)d_in[14];
  const float* Wd = (const float*)d_in[15];
  const float* bd = (const float*)d_in[16];
  const float* ln1_w = (const float*)d_in[17];
  const float* ln1_b = (const float*)d_in[18];
  const float* ln2_w = (const float*)d_in[19];
  const float* ln2_b = (const float*)d_in[20];
  const float* fnorm_w = (const float*)d_in[21];
  const float* fnorm_b = (const float*)d_in[22];
  (void)in_sizes; (void)n_in; (void)out_size; (void)ws_size;

  // workspace carve-up (~190 MB)
  char* ws = (char*)d_ws;
  size_t off = 0;
  auto alloc = [&](size_t bytes) {
    void* p = ws + off; off += (bytes + 255) & ~(size_t)255; return p;
  };
  float* x    = (float*)alloc(sizeof(float) * Mtok * Dm);       // residual
  float* cat  = (float*)alloc(sizeof(float) * Mtok * 2 * Dm);   // [normed1|ctx]
  float* nrm  = (float*)alloc(sizeof(float) * Mtok * Dm);       // normed2/final
  float* qb   = (float*)alloc(sizeof(float) * Mtok * Dm);       // q, then query
  float* kb   = (float*)alloc(sizeof(float) * Mtok * Dm);       // k, then scores
  float* vb   = (float*)alloc(sizeof(float) * Mtok * Dm);       // v, then token_recipe
  float* tbuf = (float*)alloc(sizeof(float) * Mtok * DFF_);     // t, then u
  float* hf   = (float*)alloc(sizeof(float) * Mtok * DFF_);     // gelu output
  // bf16 [N,K] weight shadows
  unsigned short* wqT = (unsigned short*)alloc(2ull * Ll * Dm * Dm);
  unsigned short* wkT = (unsigned short*)alloc(2ull * Ll * Dm * Dm);
  unsigned short* wvT = (unsigned short*)alloc(2ull * Ll * Dm * Dm);
  unsigned short* wsT = (unsigned short*)alloc(2ull * Ll * Dm * 2 * Dm);
  unsigned short* wdT = (unsigned short*)alloc(2ull * Ll * Dm * DFF_);
  unsigned short* pAT = (unsigned short*)alloc(2ull * NB_ * RANK_ * Dm);
  unsigned short* bBT = (unsigned short*)alloc(2ull * DFF_ * DFF_);
  unsigned short* teB = (unsigned short*)alloc(2ull * VOCAB_ * Dm);
  unsigned short* neB = (unsigned short*)alloc(2ull * NNEU_ * Dm);
  float* out = (float*)d_out;

  // ---- one-time prep: embedding + bf16 [N,K] weight shadows ----
  embed_kernel<<<(Mtok * Dm + 255) / 256, 256, 0, stream>>>(input_ids, token_emb, pos_emb, x);
  for (int l = 0; l < Ll; ++l) {
    convT_kernel<<<(Dm * Dm + 255) / 256, 256, 0, stream>>>(Wq + (size_t)l * Dm * Dm, wqT + (size_t)l * Dm * Dm, Dm, Dm);
    convT_kernel<<<(Dm * Dm + 255) / 256, 256, 0, stream>>>(Wk + (size_t)l * Dm * Dm, wkT + (size_t)l * Dm * Dm, Dm, Dm);
    convT_kernel<<<(Dm * Dm + 255) / 256, 256, 0, stream>>>(Wv + (size_t)l * Dm * Dm, wvT + (size_t)l * Dm * Dm, Dm, Dm);
    convT_kernel<<<(2 * Dm * Dm + 255) / 256, 256, 0, stream>>>(Ws + (size_t)l * 2 * Dm * Dm, wsT + (size_t)l * 2 * Dm * Dm, 2 * Dm, Dm);
    convT_kernel<<<(DFF_ * Dm + 255) / 256, 256, 0, stream>>>(Wd + (size_t)l * DFF_ * Dm, wdT + (size_t)l * DFF_ * Dm, DFF_, Dm);
  }
  packAT_kernel<<<(NB_ * RANK_ * Dm + 255) / 256, 256, 0, stream>>>(basis_A, pAT);
  convT_kernel<<<(DFF_ * DFF_ + 255) / 256, 256, 0, stream>>>(basis_B, bBT, DFF_, DFF_);
  conv_kernel<<<(VOCAB_ * Dm + 255) / 256, 256, 0, stream>>>(token_emb, teB, VOCAB_ * Dm);

  dim3 gD((Dm + 127) / 128, (Mtok + 127) / 128);     // N=512 GEMMs
  dim3 gN((NNEU_ + 127) / 128, (Mtok + 127) / 128);  // N=64 GEMM
  dim3 gF((DFF_ + 127) / 128, (Mtok + 127) / 128);   // N=2048 GEMMs

  for (int l = 0; l < Ll; ++l) {
    const float* rec_l = recipe + (size_t)l * NNEU_ * NB_;
    // normed1 -> cat[:, :D]
    ln_kernel<<<Mtok, 128, 0, stream>>>(x, ln1_w + l * Dm, ln1_b + l * Dm, cat, 2 * Dm);
    // QKV projections
    gemm_tdm<0, false><<<gD, 256, 0, stream>>>(cat, 2 * Dm, wqT + (size_t)l * Dm * Dm, bq + l * Dm, nullptr, qb, Mtok, Dm, Dm);
    gemm_tdm<0, false><<<gD, 256, 0, stream>>>(cat, 2 * Dm, wkT + (size_t)l * Dm * Dm, bk + l * Dm, nullptr, kb, Mtok, Dm, Dm);
    gemm_tdm<0, false><<<gD, 256, 0, stream>>>(cat, 2 * Dm, wvT + (size_t)l * Dm * Dm, bv + l * Dm, nullptr, vb, Mtok, Dm, Dm);
    // fused causal attention -> ctx = cat[:, D:]
    attn_kernel<<<dim3(Sq / 64, Bsz * Hh), 128, 0, stream>>>(qb, kb, vb, cat + Dm, 2 * Dm);
    // query = [normed1|ctx] @ Ws + bscore   (qb is free now)
    gemm_tdm<0, false><<<gD, 256, 0, stream>>>(cat, 2 * Dm, wsT + (size_t)l * 2 * Dm * Dm, bscore + l * Dm, nullptr, qb, Mtok, Dm, 2 * Dm);
    // neuron_emb (bf16 [64,512]), scores = query @ neuron_emb^T  (kb free now)
    nemb_kernel<<<NNEU_, 128, 0, stream>>>(rec_l, basis_emb, neB);
    gemm_tdm<0, false><<<gN, 256, 0, stream>>>(qb, Dm, neB, nullptr, nullptr, kb, Mtok, NNEU_, Dm);
    // router: top-8 + softmax + recipe mixing -> token_recipe in vb
    router_kernel<<<(Mtok + 255) / 256, 256, 0, stream>>>(kb, rec_l, vb, Mtok);
    // recipe FFN
    ln_kernel<<<Mtok, 128, 0, stream>>>(x, ln2_w + l * Dm, ln2_b + l * Dm, nrm, Dm);
    gemm_tdm<0, false><<<gF, 256, 0, stream>>>(nrm, Dm, pAT, nullptr, nullptr, tbuf, Mtok, DFF_, Dm);
    mix_kernel<<<Mtok, 64, 0, stream>>>(tbuf, vb);           // t -> u in place
    gemm_tdm<1, false><<<gF, 256, 0, stream>>>(tbuf, DFF_, bBT, nullptr, nullptr, hf, Mtok, DFF_, DFF_);
    // x = x + hf @ Wd + bd
    gemm_tdm<0, true><<<gD, 256, 0, stream>>>(hf, DFF_, wdT + (size_t)l * DFF_ * Dm, bd + l * Dm, x, x, Mtok, Dm, DFF_);
  }

  // final norm + tied head: out = xn @ token_emb^T
  ln_kernel<<<Mtok, 128, 0, stream>>>(x, fnorm_w, fnorm_b, nrm, Dm);
  dim3 gh((VOCAB_ + 127) / 128, (Mtok + 127) / 128);
  gemm_tdm<0, false><<<gh, 256, 0, stream>>>(nrm, Dm, teB, nullptr, nullptr, out, Mtok, VOCAB_, Dm);
}
